// Lstm_77163382440447
// MI455X (gfx1250) — matrix operand order repaired and checked by reading, not compiled, not run on hardware
//
#include <hip/hip_runtime.h>

// ---------------------------------------------------------------------------
// MI455X (gfx1250, wave32) implementation of a 2-layer bi-LSTM + MLP head.
//
// Roofline: ~75 GFLOP total, dominated by a T=1024 *sequential* recurrence
// (latency-bound, not BW/compute bound).  Strategy:
//   * v_wmma_f32_16x16x32_f16 for all per-step matmuls (f16 in, f32 accum).
//   * Layer-1 weight fragments pinned in VGPRs for all 1024 steps;
//     layer-2 weights pinned in LDS (51 KB).  h in LDS, c in registers.
//   * Each wave owns a 16-wide hidden slice across all 4 gates -> the LSTM
//     pointwise update is lane-local (no shuffles).
//   * Activations: hardware v_tanh_f32 (gfx1250 TRANS op) when available;
//     sigmoid(x) = 0.5 + 0.5*tanh(x/2) -> 1 trans + 1 fma, no IEEE divide
//     on the recurrence critical path.
//   * Activations staged f32 in d_ws (hs1: 294 MB, hs2: 88 MB), streamed
//     once by the fused MLP head (~16 us at 23.3 TB/s).
// ---------------------------------------------------------------------------

typedef _Float16 v16h __attribute__((ext_vector_type(16)));
typedef _Float16 v8h  __attribute__((ext_vector_type(8)));
typedef float    v8f  __attribute__((ext_vector_type(8)));

#define NB    512
#define NT    1024
#define NF    8
#define H1    70
#define H1P   80      // per-gate padded width, layer 1 (5 tiles of 16)
#define HS1W  140     // 2*H1 (layer-1 bi output width)
#define H2    21
#define H2P   32      // per-gate padded width, layer 2 (2 tiles of 16)
#define HS2W  42      // 2*H2
#define G2P   128     // 4*H2P
#define K2TOT 192     // layer-2 K: 160 (input pad of 140) + 32 (h pad of 21)
#define KP2   200     // padded K stride for layer-2 W in LDS (f16 units)
#define HLP1  104     // padded h row stride, layer 1 (>=96, mult of 8)
#define HLP2  40      // padded h row stride, layer 2 (>=32, mult of 8)
#define ND1   30
#define ND2   20

// ---- Activations: keep the recurrence critical path on TRANS ops ----------
__device__ __forceinline__ float tanh_f(float x) {
#if __has_builtin(__builtin_amdgcn_tanhf)
    return __builtin_amdgcn_tanhf(x);            // v_tanh_f32 (gfx1250)
#elif __has_builtin(__builtin_amdgcn_tanh_f32)
    return __builtin_amdgcn_tanh_f32(x);
#else
    // 1 - 2*rcp(e^{2x}+1): v_exp_f32 + v_rcp_f32, no IEEE divide chain.
    float e = __expf(2.0f * x);
    return fmaf(-2.0f, __builtin_amdgcn_rcpf(e + 1.0f), 1.0f);
#endif
}
__device__ __forceinline__ float sigmoid_f(float x) {
    return fmaf(0.5f, tanh_f(0.5f * x), 0.5f);   // 1 trans + 1 fma
}

// Load a 16-f16 fragment as two 16-byte LDS/global reads, `gap` f16 apart.
// gap = 8  -> 16 consecutive K values (unused; B fragments use gap = 16 too)
// gap = 16 -> two 8-element runs 16 apart (A-matrix lane layout)
__device__ __forceinline__ v16h ld_frag(const _Float16* p, int gap) {
    union { v16h v; v8h h[2]; } u;
    u.h[0] = *(const v8h*)(p);
    u.h[1] = *(const v8h*)(p + gap);
    return u.v;
}

// ===========================================================================
// Layer-1 bi-LSTM.  Grid: (32 batch tiles, 2 directions).  Block: 160 thr
// (5 waves).  Wave w owns hidden cols [16w,16w+16) of every gate; its four
// 16x16 f32 accumulators are the i/f/g/o pre-activations for that slice.
// K layout (128 = 4 chunks of 32):  [0..7]=x features, [8..31]=0,
// [32..127] = previous h (cols 0..95, 70..95 zero-padded).
// ===========================================================================
__global__ __launch_bounds__(160) void lstm1_kernel(
    const float* __restrict__ x,
    const float* __restrict__ wihF, const float* __restrict__ whhF,
    const float* __restrict__ bihF, const float* __restrict__ bhhF,
    const float* __restrict__ wihB, const float* __restrict__ whhB,
    const float* __restrict__ bihB, const float* __restrict__ bhhB,
    float* __restrict__ hs1)
{
    __shared__ __align__(16) _Float16 hl[16 * HLP1];   // previous h (f16)

    const int dir = blockIdx.y;
    const float* __restrict__ w_ih = dir ? wihB : wihF;
    const float* __restrict__ w_hh = dir ? whhB : whhF;
    const float* __restrict__ b_ih = dir ? bihB : bihF;
    const float* __restrict__ b_hh = dir ? bhhB : bhhF;

    const int tid   = threadIdx.x;
    const int lane  = tid & 31;
    const int wv    = tid >> 5;                 // 0..4
    const int mrow  = lane & 15;                // A/B lane row/col index
    const int khi   = (lane < 16) ? 0 : 16;     // unused (B fragments use ahi)
    const int ahi   = (lane < 16) ? 0 : 8;      // A fragment K offset
    const int rbase = (lane < 16) ? 0 : 8;      // C/D M base for this lane
    const int col   = (wv << 4) + mrow;         // padded hidden col 0..79
    const int r0    = blockIdx.x << 4;          // batch-tile base row
    const int dirOff = dir * H1;

    // ---- Build the 16 resident B fragments (4 gates x 4 K-chunks) --------
    v16h Bf[16];
#pragma unroll
    for (int g = 0; g < 4; ++g) {
        const int grow = 70 * g + col;          // row of w_ih / w_hh
#pragma unroll
        for (int c = 0; c < 4; ++c) {
            v16h f;
#pragma unroll
            for (int e = 0; e < 16; ++e) {
                const int k = 32 * c + ahi + ((e < 8) ? e : (8 + e)); // global K index 0..127
                float val = 0.0f;
                if (col < H1) {
                    if (k < NF) {
                        val = w_ih[grow * NF + k];
                    } else if (k >= 32) {
                        const int jj = k - 32;
                        if (jj < H1) val = w_hh[grow * H1 + jj];
                    }
                }
                f[e] = (_Float16)val;
            }
            Bf[(g << 2) + c] = f;
        }
    }
    float biasr[4];
#pragma unroll
    for (int g = 0; g < 4; ++g)
        biasr[g] = (col < H1) ? (b_ih[70 * g + col] + b_hh[70 * g + col]) : 0.0f;

    // zero the h buffer (K padding cols 80..95 stay zero forever)
    for (int i = tid; i < 16 * HLP1; i += 160) hl[i] = (_Float16)0.0f;
    __syncthreads();

    v8f cc = {};                                // cell state (f32, registers)

    for (int step = 0; step < NT; ++step) {
        const int t = dir ? (NT - 1 - step) : step;

        // ---- A fragments: chunk 0 = x_t, chunks 1..3 = previous h --------
        v16h A0;
#pragma unroll
        for (int e = 0; e < 16; ++e) A0[e] = (_Float16)0.0f;
        if (lane < 16) {
            const float* xr = x + ((size_t)(r0 + mrow) * NT + t) * NF;
            const float4 xa = *(const float4*)(xr);
            const float4 xb = *(const float4*)(xr + 4);
            A0[0] = (_Float16)xa.x; A0[1] = (_Float16)xa.y;
            A0[2] = (_Float16)xa.z; A0[3] = (_Float16)xa.w;
            A0[4] = (_Float16)xb.x; A0[5] = (_Float16)xb.y;
            A0[6] = (_Float16)xb.z; A0[7] = (_Float16)xb.w;
            // prefetch next timestep's x row into cache (global_prefetch_b8)
            const int tn = dir ? (t > 0 ? t - 1 : 0) : (t < NT - 1 ? t + 1 : t);
            __builtin_prefetch(x + ((size_t)(r0 + mrow) * NT + tn) * NF, 0, 1);
        }
        const _Float16* hrow = &hl[mrow * HLP1 + ahi];
        v16h A1 = ld_frag(hrow +  0, 16);
        v16h A2 = ld_frag(hrow + 32, 16);
        v16h A3 = ld_frag(hrow + 64, 16);
        __syncthreads();   // all waves done reading h before anyone rewrites

        // ---- 16 WMMAs: 4 gate tiles x 4 K chunks -------------------------
        v8f accs[4];
#pragma unroll
        for (int g = 0; g < 4; ++g) {
            v8f a;
#pragma unroll
            for (int v = 0; v < 8; ++v) a[v] = biasr[g];
            a = __builtin_amdgcn_wmma_f32_16x16x32_f16(false, A0, false, Bf[(g << 2) + 0], (short)0, a, false, false);
            a = __builtin_amdgcn_wmma_f32_16x16x32_f16(false, A1, false, Bf[(g << 2) + 1], (short)0, a, false, false);
            a = __builtin_amdgcn_wmma_f32_16x16x32_f16(false, A2, false, Bf[(g << 2) + 2], (short)0, a, false, false);
            a = __builtin_amdgcn_wmma_f32_16x16x32_f16(false, A3, false, Bf[(g << 2) + 3], (short)0, a, false, false);
            accs[g] = a;
        }

        // ---- Pointwise LSTM cell (lane-local: same (M,N) in all 4 tiles) -
#pragma unroll
        for (int v = 0; v < 8; ++v) {
            const float ci = sigmoid_f(accs[1][v]) * cc[v]
                           + sigmoid_f(accs[0][v]) * tanh_f(accs[2][v]);
            cc[v] = ci;
            const float hv = sigmoid_f(accs[3][v]) * tanh_f(ci);
            const int row = rbase + v;
            hl[row * HLP1 + col] = (_Float16)hv;
            if (col < H1)
                hs1[((size_t)(r0 + row) * NT + t) * HS1W + dirOff + col] = hv;
        }
        __syncthreads();   // h fully rewritten before next step reads it
    }
}

// ===========================================================================
// Layer-2 bi-LSTM.  Grid: (32, 2).  Block: 64 thr (2 waves).  Weights in LDS
// (128 x KP2 f16 = 50 KB).  K layout (192 = 6 chunks of 32):
// [0..159] = layer-1 features (140 valid), [160..191] = previous h (21 valid).
// ===========================================================================
__global__ __launch_bounds__(64) void lstm2_kernel(
    const float* __restrict__ hs1,
    const float* __restrict__ wihF, const float* __restrict__ whhF,
    const float* __restrict__ bihF, const float* __restrict__ bhhF,
    const float* __restrict__ wihB, const float* __restrict__ whhB,
    const float* __restrict__ bihB, const float* __restrict__ bhhB,
    float* __restrict__ hs2)
{
    __shared__ __align__(16) _Float16 W2l[G2P * KP2];  // 51200 B
    __shared__ __align__(16) _Float16 h2l[16 * HLP2];  // 1280 B

    const int dir = blockIdx.y;
    const float* __restrict__ w_ih = dir ? wihB : wihF;
    const float* __restrict__ w_hh = dir ? whhB : whhF;
    const float* __restrict__ b_ih = dir ? bihB : bihF;
    const float* __restrict__ b_hh = dir ? bhhB : bhhF;

    const int tid   = threadIdx.x;
    const int lane  = tid & 31;
    const int wv    = tid >> 5;                 // 0..1
    const int mrow  = lane & 15;
    const int khi   = (lane < 16) ? 0 : 16;
    const int ahi   = (lane < 16) ? 0 : 8;
    const int rbase = (lane < 16) ? 0 : 8;
    const int col   = (wv << 4) + mrow;         // padded hidden col 0..31
    const int r0    = blockIdx.x << 4;

    // ---- Stage padded f16 weights into LDS -------------------------------
    for (int i = tid; i < G2P * K2TOT; i += 64) {
        const int n = i / K2TOT, k = i % K2TOT;
        const int g = n >> 5, j = n & 31;
        float val = 0.0f;
        if (j < H2) {
            const int grow = H2 * g + j;
            if (k < 160) {
                if (k < HS1W) val = w_ih[grow * HS1W + k];
            } else {
                const int jj = k - 160;
                if (jj < H2) val = w_hh[grow * H2 + jj];
            }
        }
        W2l[n * KP2 + k] = (_Float16)val;
    }
    for (int i = tid; i < 16 * HLP2; i += 64) h2l[i] = (_Float16)0.0f;

    float biasr[4];
#pragma unroll
    for (int g = 0; g < 4; ++g)
        biasr[g] = (col < H2) ? (b_ih[H2 * g + col] + b_hh[H2 * g + col]) : 0.0f;
    __syncthreads();

    v8f cc = {};

    for (int step = 0; step < NT; ++step) {
        const int t = dir ? (NT - 1 - step) : step;

        // ---- A: 5 input chunks from hs1 (f32 -> f16) + 1 h chunk ---------
        const float* row = hs1 + ((size_t)(r0 + mrow) * NT + t) * HS1W;
        v16h A[6];
#pragma unroll
        for (int c = 0; c < 4; ++c) {          // cols < 128: no guard needed
            const int kb = 32 * c + ahi;
            const float4 a0 = *(const float4*)(row + kb);
            const float4 a1 = *(const float4*)(row + kb + 4);
            const float4 b0 = *(const float4*)(row + kb + 16);
            const float4 b1 = *(const float4*)(row + kb + 20);
            A[c][0]  = (_Float16)a0.x; A[c][1]  = (_Float16)a0.y;
            A[c][2]  = (_Float16)a0.z; A[c][3]  = (_Float16)a0.w;
            A[c][4]  = (_Float16)a1.x; A[c][5]  = (_Float16)a1.y;
            A[c][6]  = (_Float16)a1.z; A[c][7]  = (_Float16)a1.w;
            A[c][8]  = (_Float16)b0.x; A[c][9]  = (_Float16)b0.y;
            A[c][10] = (_Float16)b0.z; A[c][11] = (_Float16)b0.w;
            A[c][12] = (_Float16)b1.x; A[c][13] = (_Float16)b1.y;
            A[c][14] = (_Float16)b1.z; A[c][15] = (_Float16)b1.w;
        }
        {
            const int kb = 128 + ahi;           // chunk 4: guarded (140 edge)
#pragma unroll
            for (int e = 0; e < 16; ++e) {
                const int cidx = kb + ((e < 8) ? e : (16 + e - 8));
                A[4][e] = (_Float16)((cidx < HS1W) ? row[cidx] : 0.0f);
            }
        }
        A[5] = ld_frag(&h2l[mrow * HLP2 + ahi], 16);
        __syncthreads();

        // ---- 24 WMMAs: 4 gate tiles x 6 K chunks (B streamed from LDS) ---
        v8f accs[4];
#pragma unroll
        for (int g = 0; g < 4; ++g) {
            const int nt = 2 * g + wv;          // this wave's tile of gate g
            const _Float16* wb = &W2l[(nt * 16 + mrow) * KP2 + ahi];
            v8f a;
#pragma unroll
            for (int v = 0; v < 8; ++v) a[v] = biasr[g];
#pragma unroll
            for (int c = 0; c < 6; ++c) {
                const v16h B = ld_frag(wb + 32 * c, 16);
                a = __builtin_amdgcn_wmma_f32_16x16x32_f16(false, A[c], false, B, (short)0, a, false, false);
            }
            accs[g] = a;
        }

        // ---- LSTM cell + write h ----------------------------------------
#pragma unroll
        for (int v = 0; v < 8; ++v) {
            const float ci = sigmoid_f(accs[1][v]) * cc[v]
                           + sigmoid_f(accs[0][v]) * tanh_f(accs[2][v]);
            cc[v] = ci;
            const float hv = sigmoid_f(accs[3][v]) * tanh_f(ci);
            const int rowi = rbase + v;
            h2l[rowi * HLP2 + col] = (_Float16)hv;
            if (col < H2)
                hs2[((size_t)(r0 + rowi) * NT + t) * HS2W + dir * H2 + col] = hv;
        }
        __syncthreads();
    }
}

// ===========================================================================
// Fused MLP head: relu(h@d1^T+b1) -> relu(@d2^T+b2) -> @out^T+b.
// One thread per (b,t) row; weights cached in LDS.  Streams hs2 once.
// ===========================================================================
__global__ __launch_bounds__(256) void head_kernel(
    const float* __restrict__ hs2,
    const float* __restrict__ d1w, const float* __restrict__ d1b,
    const float* __restrict__ d2w, const float* __restrict__ d2b,
    const float* __restrict__ ow,  const float* __restrict__ ob,
    float* __restrict__ out)
{
    __shared__ float s1w[ND1 * HS2W];
    __shared__ float s1b[ND1];
    __shared__ float s2w[ND2 * ND1];
    __shared__ float s2b[ND2];
    __shared__ float sow[ND2];
    __shared__ float sob;

    const int tid = threadIdx.x;
    for (int i = tid; i < ND1 * HS2W; i += 256) s1w[i] = d1w[i];
    for (int i = tid; i < ND2 * ND1;  i += 256) s2w[i] = d2w[i];
    if (tid < ND1) s1b[tid] = d1b[tid];
    if (tid < ND2) { s2b[tid] = d2b[tid]; sow[tid] = ow[tid]; }
    if (tid == 0) sob = ob[0];
    __syncthreads();

    const size_t idx = (size_t)blockIdx.x * 256 + tid;   // exact: 2048*256
    const float* h = hs2 + idx * HS2W;

    float v[HS2W];
#pragma unroll
    for (int j = 0; j < HS2W; ++j) v[j] = h[j];

    float a1[ND1];
    for (int i = 0; i < ND1; ++i) {
        float s = s1b[i];
#pragma unroll
        for (int j = 0; j < HS2W; ++j) s += s1w[i * HS2W + j] * v[j];
        a1[i] = fmaxf(s, 0.0f);
    }
    float a2[ND2];
    for (int i = 0; i < ND2; ++i) {
        float s = s2b[i];
#pragma unroll
        for (int j = 0; j < ND1; ++j) s += s2w[i * ND1 + j] * a1[j];
        a2[i] = fmaxf(s, 0.0f);
    }
    float y = sob;
#pragma unroll
    for (int i = 0; i < ND2; ++i) y += sow[i] * a2[i];
    out[idx] = y;
}

// ===========================================================================
extern "C" void kernel_launch(void* const* d_in, const int* in_sizes, int n_in,
                              void* d_out, int out_size, void* d_ws, size_t ws_size,
                              hipStream_t stream)
{
    (void)in_sizes; (void)n_in; (void)out_size; (void)ws_size;

    const float* x      = (const float*)d_in[0];
    const float* wih1f  = (const float*)d_in[1];
    const float* whh1f  = (const float*)d_in[2];
    const float* bih1f  = (const float*)d_in[3];
    const float* bhh1f  = (const float*)d_in[4];
    const float* wih1b  = (const float*)d_in[5];
    const float* whh1b  = (const float*)d_in[6];
    const float* bih1b  = (const float*)d_in[7];
    const float* bhh1b  = (const float*)d_in[8];
    const float* wih2f  = (const float*)d_in[9];
    const float* whh2f  = (const float*)d_in[10];
    const float* bih2f  = (const float*)d_in[11];
    const float* bhh2f  = (const float*)d_in[12];
    const float* wih2b  = (const float*)d_in[13];
    const float* whh2b  = (const float*)d_in[14];
    const float* bih2b  = (const float*)d_in[15];
    const float* bhh2b  = (const float*)d_in[16];
    const float* d1w    = (const float*)d_in[17];
    const float* d1b    = (const float*)d_in[18];
    const float* d2w    = (const float*)d_in[19];
    const float* d2b    = (const float*)d_in[20];
    const float* outw   = (const float*)d_in[21];
    const float* outb   = (const float*)d_in[22];

    // Workspace layout (f32): hs1 (B,T,140) = 294 MB, hs2 (B,T,42) = 88 MB.
    float* hs1 = (float*)d_ws;
    float* hs2 = hs1 + (size_t)NB * NT * HS1W;

    lstm1_kernel<<<dim3(NB / 16, 2), 160, 0, stream>>>(
        x, wih1f, whh1f, bih1f, bhh1f, wih1b, whh1b, bih1b, bhh1b, hs1);

    lstm2_kernel<<<dim3(NB / 16, 2), 64, 0, stream>>>(
        hs1, wih2f, whh2f, bih2f, bhh2f, wih2b, whh2b, bih2b, bhh2b, hs2);

    head_kernel<<<(NB * NT) / 256, 256, 0, stream>>>(
        hs2, d1w, d1b, d2w, d2b, outw, outb, (float*)d_out);
}
